// AxialDecoder_64759516889847
// MI455X (gfx1250) — matrix operand order repaired and checked by reading, not compiled, not run on hardware
//
#include <hip/hip_runtime.h>
#include <math.h>

typedef _Float16 half_t;
typedef __attribute__((ext_vector_type(16))) _Float16 v16h;
typedef __attribute__((ext_vector_type(8)))  _Float16 v8h;
typedef __attribute__((ext_vector_type(8)))  float    v8f;

#define B_   32
#define S_   64
#define E_   256
#define H_   4
#define W_   5
#define T_   (B_*S_*H_*W_)   /* 40960 tokens */
#define HW_  (H_*W_)         /* 20 */
#define SHW_ (S_*H_*W_)      /* 1280 */
#define SCALE_ 0.25f         /* 16^-0.5 */

// ---------------------------------------------------------------------------
// NULL-tensor TDM probe: all-zero descriptor groups -> D#.count==0 == NULL
// tensor (documented NOP). Exercises TENSOR_LOAD_TO_LDS + s_wait_tensorcnt
// without moving data. Arity differs between toolchains (bridge doc):
// clang-22/ROCm7.2 = 5 args, clang-23/amdgpu-toolchain = 6 args.
// ---------------------------------------------------------------------------
__global__ void k_tdm_nop()
{
    __shared__ int lds_dummy[16];
    lds_dummy[threadIdx.x & 15] = 0;   // keep LDS allocated
    typedef __attribute__((ext_vector_type(4))) unsigned int v4u;
    typedef __attribute__((ext_vector_type(8))) int v8i_;
    typedef __attribute__((ext_vector_type(4))) int v4i_;
    v4u  g0 = {};
    v8i_ g1 = {};
    v4i_ g2 = {};
    v4i_ g3 = {};
#if defined(__clang_major__) && (__clang_major__ >= 23)
    v8i_ g4 = {};
    __builtin_amdgcn_tensor_load_to_lds(g0, g1, g2, g3, g4, 0);
#else
    __builtin_amdgcn_tensor_load_to_lds(g0, g1, g2, g3, 0);
#endif
    __builtin_amdgcn_s_wait_tensorcnt(0);
}

// ---------------------------------------------------------------------------
// Fused positional add + (B,S,E,H,W) -> token-major (t, E) transpose.
// Emits fp32 master and f16 copy for the WMMA GEMMs.
// ---------------------------------------------------------------------------
__global__ __launch_bounds__(256) void k_addpos(const float* __restrict__ x,
                                                const float* __restrict__ ps,
                                                const float* __restrict__ ph,
                                                const float* __restrict__ pw,
                                                float* __restrict__ X32,
                                                half_t* __restrict__ X16)
{
    int gid = blockIdx.x * 256 + threadIdx.x;   // over T_*E_
    int t = gid >> 8, e = gid & 255;
    int w  = t % W_;  int t2 = t / W_;
    int h  = t2 % H_; int t3 = t2 / H_;
    int s  = t3 % S_; int b  = t3 / S_;
    size_t xi = ((((size_t)b * S_ + s) * E_ + e) * H_ + h) * W_ + w;
    float v = x[xi] + ps[s * E_ + e] + ph[e * H_ + h] + pw[e * W_ + w];
    X32[gid] = v;
    X16[gid] = (half_t)v;
}

// fp32 -> f16 convert
__global__ __launch_bounds__(256) void k_cvt16(const float* __restrict__ src,
                                               half_t* __restrict__ dst, int n)
{
    int gid = blockIdx.x * 256 + threadIdx.x;
    if (gid < n) dst[gid] = (half_t)src[gid];
}

// ---------------------------------------------------------------------------
// A-fragment load (16-bit A 16x32, this code's contiguous K order; B uses the same):
// lane<16 -> row m, K {0..15}; lanes 16-31 -> K {16..31}. Two 16B loads.
// ---------------------------------------------------------------------------
__device__ __forceinline__ v16h ld_afrag(const half_t* __restrict__ p)
{
    v8h lo  = *(const v8h*)p;
    v8h hi8 = *(const v8h*)(p + 8);
    return __builtin_shufflevector(lo, hi8,
                                   0,1,2,3,4,5,6,7,8,9,10,11,12,13,14,15);
}

__device__ __forceinline__ v8f wmma_f16(v16h a, v16h b, v8f c)
{
    return __builtin_amdgcn_wmma_f32_16x16x32_f16(false, a, false, b,
                                                  (short)0, c, false, false);
}

// ---------------------------------------------------------------------------
// WMMA GEMM: C[M,N] = A[M,K] @ W[N,K]^T   (A,W f16 row-major, C f16)
// One wave computes 32 rows x 64 cols: 2 A frags x 4 B frags -> 8 WMMAs per
// K-step on 12 b128 loads (B reused across both row tiles).
// Block = 256 threads = 8 waves (2 row-groups x 4 col-groups) = 64x256 tile.
// ---------------------------------------------------------------------------
__global__ __launch_bounds__(256) void k_gemm_f16(const half_t* __restrict__ A,
                                                  const half_t* __restrict__ Wt,
                                                  half_t* __restrict__ C,
                                                  int M, int N, int K)
{
    const int lane  = threadIdx.x & 31;
    const int wave  = threadIdx.x >> 5;
    const int wr    = wave >> 2, wc = wave & 3;
    const int m0    = blockIdx.x * 64 + wr * 32;
    const int nbase = blockIdx.y * 256 + wc * 64;
    const int mr    = lane & 15, hi = lane >> 4;

    v8f acc0[4] = {};
    v8f acc1[4] = {};
    const half_t* arow0 = A + (size_t)(m0 + mr) * K + hi * 16;
    const half_t* arow1 = arow0 + (size_t)16 * K;
    for (int kb = 0; kb < K; kb += 32) {
        v16h a0 = ld_afrag(arow0 + kb);
        v16h a1 = ld_afrag(arow1 + kb);
#pragma unroll
        for (int sub = 0; sub < 4; ++sub) {
            // B frag: lane -> col n, 16 contiguous K per lane (lo lanes K 0-15, hi 16-31)
            const half_t* brow = Wt + (size_t)(nbase + sub * 16 + mr) * K + kb + hi * 16;
            v16h b = *(const v16h*)brow;
            acc0[sub] = wmma_f16(a0, b, acc0[sub]);
            acc1[sub] = wmma_f16(a1, b, acc1[sub]);
        }
    }
    const int crow = m0 + hi * 8;        // C/D layout: VGPR r -> row r + 8*hi
#pragma unroll
    for (int sub = 0; sub < 4; ++sub) {
        int n = nbase + sub * 16 + mr;
#pragma unroll
        for (int r = 0; r < 8; ++r) {
            C[(size_t)(crow + r) * N + n]      = (half_t)acc0[sub][r];
            C[(size_t)(crow + 16 + r) * N + n] = (half_t)acc1[sub][r];
        }
    }
}

// Same tile loop, fp32 output with bias + cross-axis accumulation.
__global__ __launch_bounds__(256) void k_gemm_acc(const half_t* __restrict__ A,
                                                  const half_t* __restrict__ Wt,
                                                  const float* __restrict__ bias,
                                                  float* __restrict__ C,
                                                  int M, int N, int K, int init)
{
    const int lane  = threadIdx.x & 31;
    const int wave  = threadIdx.x >> 5;
    const int wr    = wave >> 2, wc = wave & 3;
    const int m0    = blockIdx.x * 64 + wr * 32;
    const int nbase = blockIdx.y * 256 + wc * 64;
    const int mr    = lane & 15, hi = lane >> 4;

    v8f acc0[4] = {};
    v8f acc1[4] = {};
    const half_t* arow0 = A + (size_t)(m0 + mr) * K + hi * 16;
    const half_t* arow1 = arow0 + (size_t)16 * K;
    for (int kb = 0; kb < K; kb += 32) {
        v16h a0 = ld_afrag(arow0 + kb);
        v16h a1 = ld_afrag(arow1 + kb);
#pragma unroll
        for (int sub = 0; sub < 4; ++sub) {
            const half_t* brow = Wt + (size_t)(nbase + sub * 16 + mr) * K + kb + hi * 16;
            v16h b = *(const v16h*)brow;
            acc0[sub] = wmma_f16(a0, b, acc0[sub]);
            acc1[sub] = wmma_f16(a1, b, acc1[sub]);
        }
    }
    const int crow = m0 + hi * 8;
#pragma unroll
    for (int sub = 0; sub < 4; ++sub) {
        int n = nbase + sub * 16 + mr;
        float bv = bias[n];
#pragma unroll
        for (int r = 0; r < 8; ++r) {
            size_t ci0 = (size_t)(crow + r) * N + n;
            size_t ci1 = (size_t)(crow + 16 + r) * N + n;
            float v0 = acc0[sub][r] + bv;
            float v1 = acc1[sub][r] + bv;
            if (!init) { v0 += C[ci0]; v1 += C[ci1]; }
            C[ci0] = v0;
            C[ci1] = v1;
        }
    }
}

// ---------------------------------------------------------------------------
// Axial attention core. One block = one (sequence, head); K/V staged in LDS;
// online softmax in fp32. t_len = 64 (seq axis) / 4 (H) / 5 (W).
// ---------------------------------------------------------------------------
__global__ __launch_bounds__(64) void k_attn(const half_t* __restrict__ Q,
                                             const half_t* __restrict__ KV,
                                             half_t* __restrict__ O, int axis)
{
    __shared__ float ks[64 * 16];
    __shared__ float vs[64 * 16];
    const int head = blockIdx.x & 15;
    const int sq   = blockIdx.x >> 4;
    int base, stride, tl;
    if (axis == 0)      { base = (sq / HW_) * SHW_ + (sq % HW_);     stride = HW_; tl = S_; }
    else if (axis == 1) { base = (sq / W_) * HW_ + (sq % W_);        stride = W_;  tl = H_; }
    else                { base = (sq / H_) * HW_ + (sq % H_) * W_;   stride = 1;   tl = W_; }

    const int tid = threadIdx.x;
    for (int idx = tid; idx < tl * 16; idx += 64) {
        int s = idx >> 4, d = idx & 15;
        size_t row = (size_t)(base + s * stride);
        ks[idx] = (float)KV[row * 512 + head * 16 + d];
        vs[idx] = (float)KV[row * 512 + 256 + head * 16 + d];
    }
    __syncthreads();

    if (tid < tl) {
        size_t row = (size_t)(base + tid * stride);
        float q[16];
#pragma unroll
        for (int d = 0; d < 16; ++d) q[d] = (float)Q[row * 256 + head * 16 + d];
        float m = -3.0e38f, l = 0.f, o[16];
#pragma unroll
        for (int d = 0; d < 16; ++d) o[d] = 0.f;
        for (int s = 0; s < tl; ++s) {
            float dot = 0.f;
#pragma unroll
            for (int d = 0; d < 16; ++d) dot += q[d] * ks[s * 16 + d];
            dot *= SCALE_;
            float mn   = fmaxf(m, dot);
            float corr = __expf(m - mn);
            float p    = __expf(dot - mn);
            l = l * corr + p;
#pragma unroll
            for (int d = 0; d < 16; ++d) o[d] = o[d] * corr + p * vs[s * 16 + d];
            m = mn;
        }
        float inv = 1.f / l;
#pragma unroll
        for (int d = 0; d < 16; ++d)
            O[row * 256 + head * 16 + d] = (half_t)(o[d] * inv);
    }
}

// ---------------------------------------------------------------------------
// Decode: one wave per token, 256-dot + wave reduction + sigmoid.
// ---------------------------------------------------------------------------
__global__ __launch_bounds__(256) void k_decode(const float* __restrict__ X32,
                                                const float* __restrict__ dw,
                                                const float* __restrict__ db,
                                                float* __restrict__ out)
{
    const int lane = threadIdx.x & 31;
    const int t    = blockIdx.x * 8 + (threadIdx.x >> 5);
    float sum = 0.f;
#pragma unroll
    for (int j = 0; j < 8; ++j) {
        int e = lane + j * 32;
        sum += X32[(size_t)t * 256 + e] * dw[e];
    }
#pragma unroll
    for (int off = 16; off > 0; off >>= 1)
        sum += __shfl_xor(sum, off, 32);
    if (lane == 0) out[t] = 1.f / (1.f + __expf(-(sum + db[0])));
}

// ---------------------------------------------------------------------------
extern "C" void kernel_launch(void* const* d_in, const int* in_sizes, int n_in,
                              void* d_out, int out_size, void* d_ws, size_t ws_size,
                              hipStream_t stream)
{
    const float* x   = (const float*)d_in[0];
    const float* ps  = (const float*)d_in[1];
    const float* ph  = (const float*)d_in[2];
    const float* pw  = (const float*)d_in[3];
    const float* wq  = (const float*)d_in[4];   // (2,3,256,256)
    const float* wkv = (const float*)d_in[5];   // (2,3,512,256)
    const float* wow = (const float*)d_in[6];   // (2,3,256,256)
    const float* wob = (const float*)d_in[7];   // (2,3,256)
    const float* dw  = (const float*)d_in[8];   // (1,256)
    const float* db  = (const float*)d_in[9];   // (1,)
    float* out = (float*)d_out;

    // bump allocator over d_ws (~192 MB total)
    char* p = (char*)d_ws;
    auto alloc = [&](size_t bytes) -> char* {
        char* r = p; p += (bytes + 255) & ~(size_t)255; return r;
    };
    float*  Xa    = (float*) alloc((size_t)T_ * E_ * 4);
    float*  Xb    = (float*) alloc((size_t)T_ * E_ * 4);
    half_t* X16   = (half_t*)alloc((size_t)T_ * E_ * 2);
    half_t* Q16   = (half_t*)alloc((size_t)T_ * E_ * 2);
    half_t* KV16  = (half_t*)alloc((size_t)T_ * 512 * 2);
    half_t* O16   = (half_t*)alloc((size_t)T_ * E_ * 2);
    half_t* wq16  = (half_t*)alloc((size_t)6 * 256 * 256 * 2);
    half_t* wkv16 = (half_t*)alloc((size_t)6 * 512 * 256 * 2);
    half_t* wo16  = (half_t*)alloc((size_t)6 * 256 * 256 * 2);

    // NULL-tensor TDM probe (documented NOP; exercises the TDM path)
    k_tdm_nop<<<1, 32, 0, stream>>>();

    // weights -> f16 (6 MB, once per launch, deterministic)
    k_cvt16<<<(6 * 256 * 256) / 256, 256, 0, stream>>>(wq,  wq16,  6 * 256 * 256);
    k_cvt16<<<(6 * 512 * 256) / 256, 256, 0, stream>>>(wkv, wkv16, 6 * 512 * 256);
    k_cvt16<<<(6 * 256 * 256) / 256, 256, 0, stream>>>(wow, wo16,  6 * 256 * 256);

    // positional add + transpose to token-major
    k_addpos<<<(T_ * E_) / 256, 256, 0, stream>>>(x, ps, ph, pw, Xa, X16);

    float* Xcur = Xa; float* Xnxt = Xb;
    const int nseq[3] = { B_ * H_ * W_, B_ * S_ * W_, B_ * S_ * H_ };  // 640, 10240, 8192

    for (int l = 0; l < 2; ++l) {
        for (int a = 0; a < 3; ++a) {
            const int wi = l * 3 + a;
            k_gemm_f16<<<dim3(T_ / 64, 1), 256, 0, stream>>>(
                X16, wq16 + (size_t)wi * 256 * 256, Q16, T_, 256, 256);
            k_gemm_f16<<<dim3(T_ / 64, 2), 256, 0, stream>>>(
                X16, wkv16 + (size_t)wi * 512 * 256, KV16, T_, 512, 256);
            k_attn<<<nseq[a] * 16, 64, 0, stream>>>(Q16, KV16, O16, a);
            k_gemm_acc<<<dim3(T_ / 64, 1), 256, 0, stream>>>(
                O16, wo16 + (size_t)wi * 256 * 256, wob + (size_t)wi * 256,
                Xnxt, T_, 256, 256, (a == 0) ? 1 : 0);
        }
        float* tmp = Xcur; Xcur = Xnxt; Xnxt = tmp;
        k_cvt16<<<(T_ * E_) / 256, 256, 0, stream>>>(Xcur, X16, T_ * E_);
    }

    k_decode<<<T_ / 8, 256, 0, stream>>>(Xcur, dw, db, out);

    (void)in_sizes; (void)n_in; (void)out_size; (void)ws_size;
}
